// CausalSelfAttention_80418967650984
// MI455X (gfx1250) — matrix operand order repaired and checked by reading, not compiled, not run on hardware
//
#include <hip/hip_runtime.h>
#include <hip/hip_bf16.h>

typedef __bf16 bf16_t;
typedef __attribute__((ext_vector_type(16))) __bf16 v16bf;
typedef __attribute__((ext_vector_type(8)))  __bf16 v8bf;
typedef __attribute__((ext_vector_type(8)))  float  v8f;

#define BATCH 4
#define SEQ   2048
#define EMBED 1024
#define HEADS 16
#define HD    64
#define MROWS (BATCH * SEQ)   /* 8192 */

// ---- WMMA operand loaders -------------------------------------------------
// A tile 16x32 (MxK) from row-major [.. x ld] bf16:
//   lane: row = row0 + lane%16 ; element h -> K = (h>>3)*16 + (lane>>4)*8 + (h&7)
//   => two contiguous 8-element (16B) chunks at kb and kb+16, kb = k0 + (lane>>4)*8
__device__ __forceinline__ v16bf load_a_tile(const bf16_t* p, int ld, int row0,
                                             int k0, int lane) {
  int row = row0 + (lane & 15);
  int kb  = k0 + ((lane >> 4) << 3);
  const bf16_t* base = p + (size_t)row * ld + kb;
  v8bf lo = *(const v8bf*)(base);
  v8bf hi = *(const v8bf*)(base + 16);
  v16bf r;
#pragma unroll
  for (int i = 0; i < 8; ++i) { r[i] = lo[i]; r[i + 8] = hi[i]; }
  return r;
}

// B tile 32x16 (KxN) given B^T stored row-major as [N x ld]:
//   lane: col n = n0 + lane%16 ; element h -> K = (h>>3)*16 + (lane>>4)*8 + (h&7)
//   => two contiguous 8-element (16B) chunks at kb and kb+16, kb = k0 + (lane>>4)*8
__device__ __forceinline__ v16bf load_b_tile(const bf16_t* p, int ld, int n0,
                                             int k0, int lane) {
  int n  = n0 + (lane & 15);
  int kb = k0 + ((lane >> 4) << 3);
  const bf16_t* base = p + (size_t)n * ld + kb;
  v8bf lo = *(const v8bf*)(base);
  v8bf hi = *(const v8bf*)(base + 16);
  v16bf r;
#pragma unroll
  for (int i = 0; i < 8; ++i) { r[i] = lo[i]; r[i + 8] = hi[i]; }
  return r;
}

__device__ __forceinline__ v8f wmma_bf16(v16bf a, v16bf b, v8f c) {
  return __builtin_amdgcn_wmma_f32_16x16x32_bf16(false, a, false, b,
                                                 (short)0, c, false, false);
}

// ---- fp32 -> bf16 conversions --------------------------------------------
__global__ __launch_bounds__(256) void cvt_f32_bf16(const float* __restrict__ in,
                                                    bf16_t* __restrict__ out, int n) {
  int i = blockIdx.x * 256 + threadIdx.x;
  if (i < n) out[i] = (bf16_t)in[i];
}

// W [K=1024][N=1024] fp32 -> WT [N][K] bf16 (coalesced writes)
__global__ __launch_bounds__(256) void transpose_f32_bf16(const float* __restrict__ W,
                                                          bf16_t* __restrict__ WT) {
  int i = blockIdx.x * 256 + threadIdx.x;   // i = n*1024 + k
  int n = i >> 10, k = i & 1023;
  WT[i] = (bf16_t)W[k * EMBED + n];
}

// ---- bf16 GEMM: [8192x1024] x [1024x1024] --------------------------------
// One wave computes a 16x64 output tile (4 WMMA accumulators, A reused).
// outf != nullptr : write fp32 row-major [M x 1024] (final projection).
// else           : write bf16 into per-head layout:
//                  off = (b*16+h)*(S*HD) + s*ss + d*ds   (Q/K: ss=64,ds=1 ; Vt: ss=1,ds=2048)
__global__ __launch_bounds__(256) void gemm_bf16(const bf16_t* __restrict__ A,
                                                 const bf16_t* __restrict__ BT,
                                                 bf16_t* __restrict__ outb,
                                                 float* __restrict__ outf,
                                                 int ss, int ds) {
  int lane = threadIdx.x & 31;
  int wave = threadIdx.x >> 5;
  int mt   = blockIdx.x * 8 + wave;     // 16-row tile, 0..511
  int n0   = blockIdx.y * 64;
  int g = lane >> 4, nl = lane & 15;

  v8f acc0 = {}, acc1 = {}, acc2 = {}, acc3 = {};
  for (int kk = 0; kk < EMBED; kk += 32) {
    v16bf a = load_a_tile(A, EMBED, mt * 16, kk, lane);
    acc0 = wmma_bf16(a, load_b_tile(BT, EMBED, n0,      kk, lane), acc0);
    acc1 = wmma_bf16(a, load_b_tile(BT, EMBED, n0 + 16, kk, lane), acc1);
    acc2 = wmma_bf16(a, load_b_tile(BT, EMBED, n0 + 32, kk, lane), acc2);
    acc3 = wmma_bf16(a, load_b_tile(BT, EMBED, n0 + 48, kk, lane), acc3);
  }
  v8f accs[4] = {acc0, acc1, acc2, acc3};
#pragma unroll
  for (int nt = 0; nt < 4; ++nt) {
#pragma unroll
    for (int v = 0; v < 8; ++v) {
      float val = accs[nt][v];
      int gm  = mt * 16 + v + 8 * g;            // global row (b*S + s)
      int col = n0 + nt * 16 + nl;              // global col (h*HD + d)
      if (outf) {
        outf[(size_t)gm * EMBED + col] = val;
      } else {
        int bb = gm >> 11, s = gm & (SEQ - 1);
        int h = col >> 6,  d = col & (HD - 1);
        outb[(size_t)(bb * HEADS + h) * (SEQ * HD) +
             (size_t)s * ss + (size_t)d * ds] = (bf16_t)val;
      }
    }
  }
}

// ---- Flash attention: one wave per 16-query tile -------------------------
__global__ __launch_bounds__(128) void attn_fwd(const bf16_t* __restrict__ Qg,
                                                const bf16_t* __restrict__ Kg,
                                                const bf16_t* __restrict__ Vtg,
                                                bf16_t* __restrict__ Og) {
  __shared__ alignas(32) bf16_t Plds[4][16 * 32];  // per-wave 16x32 P staging
  int lane = threadIdx.x & 31;
  int wave = threadIdx.x >> 5;
  int qt = blockIdx.x * 4 + wave;                  // 0..127
  int h  = blockIdx.y, b = blockIdx.z;
  size_t base = (size_t)(b * HEADS + h) * (SEQ * HD);
  const bf16_t* Q  = Qg  + base;                   // [S][64]
  const bf16_t* Kh = Kg  + base;                   // [S][64]
  const bf16_t* Vt = Vtg + base;                   // [64][S]
  int g = lane >> 4, nl = lane & 15;
  bf16_t* P = &Plds[wave][0];

  v16bf qa0 = load_a_tile(Q, HD, qt * 16, 0,  lane);   // d = 0..31
  v16bf qa1 = load_a_tile(Q, HD, qt * 16, 32, lane);   // d = 32..63

  float m[8], l[8];
  v8f o0 = {}, o1 = {}, o2 = {}, o3 = {};
#pragma unroll
  for (int v = 0; v < 8; ++v) { m[v] = -3.0e38f; l[v] = 0.f; }

  const float scale = 0.125f;                     // 1/sqrt(64)
  int kend = qt * 16 + 15;
  for (int kb = 0; kb <= kend; kb += 32) {
    if (kb + 32 <= kend)
      __builtin_prefetch(Kh + (size_t)(kb + 32) * HD, 0, 1);

    // scores: two 16x16 tiles over keys kb..kb+31 (K-dim = head dim 64)
    v8f s0 = {}, s1 = {};
    s0 = wmma_bf16(qa0, load_b_tile(Kh, HD, kb,      0,  lane), s0);
    s0 = wmma_bf16(qa1, load_b_tile(Kh, HD, kb,      32, lane), s0);
    s1 = wmma_bf16(qa0, load_b_tile(Kh, HD, kb + 16, 0,  lane), s1);
    s1 = wmma_bf16(qa1, load_b_tile(Kh, HD, kb + 16, 32, lane), s1);

    float alpha[8];
#pragma unroll
    for (int v = 0; v < 8; ++v) {
      int r = v + 8 * g;
      int q = qt * 16 + r;
      float x0 = (kb + nl      <= q) ? s0[v] * scale : -3.0e38f;
      float x1 = (kb + 16 + nl <= q) ? s1[v] * scale : -3.0e38f;
      // row max across the 16 lanes of this half-wave
      float t = fmaxf(x0, x1);
      t = fmaxf(t, __shfl_xor(t, 1, 32));
      t = fmaxf(t, __shfl_xor(t, 2, 32));
      t = fmaxf(t, __shfl_xor(t, 4, 32));
      t = fmaxf(t, __shfl_xor(t, 8, 32));
      float mnew = fmaxf(m[v], t);
      alpha[v] = __expf(m[v] - mnew);
      m[v] = mnew;
      float e0 = __expf(x0 - mnew);
      float e1 = __expf(x1 - mnew);
      float rs = e0 + e1;
      rs += __shfl_xor(rs, 1, 32);
      rs += __shfl_xor(rs, 2, 32);
      rs += __shfl_xor(rs, 4, 32);
      rs += __shfl_xor(rs, 8, 32);
      l[v] = l[v] * alpha[v] + rs;
      // stage P in row-major 16x32 for the A-operand re-layout
      P[r * 32 + nl]      = (bf16_t)e0;
      P[r * 32 + 16 + nl] = (bf16_t)e1;
    }
#pragma unroll
    for (int v = 0; v < 8; ++v) {
      o0[v] *= alpha[v]; o1[v] *= alpha[v];
      o2[v] *= alpha[v]; o3[v] *= alpha[v];
    }
    asm volatile("s_wait_dscnt 0" ::: "memory");
    v16bf pa = load_a_tile(P, 32, 0, 0, lane);
    o0 = wmma_bf16(pa, load_b_tile(Vt, SEQ, 0,  kb, lane), o0);
    o1 = wmma_bf16(pa, load_b_tile(Vt, SEQ, 16, kb, lane), o1);
    o2 = wmma_bf16(pa, load_b_tile(Vt, SEQ, 32, kb, lane), o2);
    o3 = wmma_bf16(pa, load_b_tile(Vt, SEQ, 48, kb, lane), o3);
  }

  v8f os[4] = {o0, o1, o2, o3};
#pragma unroll
  for (int v = 0; v < 8; ++v) {
    float inv = 1.0f / l[v];
    int q = qt * 16 + v + 8 * g;
#pragma unroll
    for (int nt = 0; nt < 4; ++nt) {
      int d = nt * 16 + nl;
      Og[(size_t)(b * SEQ + q) * EMBED + h * HD + d] = (bf16_t)(os[nt][v] * inv);
    }
  }
}

// ---- host-side orchestration ---------------------------------------------
extern "C" void kernel_launch(void* const* d_in, const int* in_sizes, int n_in,
                              void* d_out, int out_size, void* d_ws, size_t ws_size,
                              hipStream_t stream) {
  const float* x  = (const float*)d_in[0];
  const float* Wq = (const float*)d_in[1];
  const float* Wk = (const float*)d_in[2];
  const float* Wv = (const float*)d_in[3];
  const float* Wo = (const float*)d_in[4];

  char* ws = (char*)d_ws;                          // 88 MiB used
  bf16_t* Xb   = (bf16_t*)(ws);                    // 16 MiB  x (bf16)
  bf16_t* WqT  = (bf16_t*)(ws + (16u << 20));      //  2 MiB  Wq^T
  bf16_t* WkT  = (bf16_t*)(ws + (18u << 20));      //  2 MiB  Wk^T
  bf16_t* WvT  = (bf16_t*)(ws + (20u << 20));      //  2 MiB  Wv^T
  bf16_t* WoT  = (bf16_t*)(ws + (22u << 20));      //  2 MiB  Wo^T
  bf16_t* Qb   = (bf16_t*)(ws + (24u << 20));      // 16 MiB  Q  [B,H,S,64]
  bf16_t* Kb   = (bf16_t*)(ws + (40u << 20));      // 16 MiB  K  [B,H,S,64]
  bf16_t* Vtb  = (bf16_t*)(ws + (56u << 20));      // 16 MiB  V^T [B,H,64,S]
  bf16_t* Attn = (bf16_t*)(ws + (72u << 20));      // 16 MiB  attended [B,S,E]

  const int NX = MROWS * EMBED;                    // 8388608
  cvt_f32_bf16<<<NX / 256, 256, 0, stream>>>(x, Xb, NX);
  const int NW = EMBED * EMBED;                    // 1048576
  transpose_f32_bf16<<<NW / 256, 256, 0, stream>>>(Wq, WqT);
  transpose_f32_bf16<<<NW / 256, 256, 0, stream>>>(Wk, WkT);
  transpose_f32_bf16<<<NW / 256, 256, 0, stream>>>(Wv, WvT);
  transpose_f32_bf16<<<NW / 256, 256, 0, stream>>>(Wo, WoT);

  dim3 gemmGrid(MROWS / 16 / 8, EMBED / 64);       // (64, 16)
  gemm_bf16<<<gemmGrid, 256, 0, stream>>>(Xb, WqT, Qb,  nullptr, HD, 1);
  gemm_bf16<<<gemmGrid, 256, 0, stream>>>(Xb, WkT, Kb,  nullptr, HD, 1);
  gemm_bf16<<<gemmGrid, 256, 0, stream>>>(Xb, WvT, Vtb, nullptr, 1, SEQ);

  attn_fwd<<<dim3(SEQ / 16 / 4, HEADS, BATCH), 128, 0, stream>>>(Qb, Kb, Vtb, Attn);

  gemm_bf16<<<gemmGrid, 256, 0, stream>>>(Attn, WoT, nullptr, (float*)d_out, 0, 0);
}
